// DecoderLayer_17789754540257
// MI455X (gfx1250) — matrix operand order repaired and checked by reading, not compiled, not run on hardware
//
#include <hip/hip_runtime.h>
#include <hip/hip_bf16.h>
#include <math.h>

// ---------------- CDNA5 (gfx1250) wave32 WMMA transformer decoder layer ----
// B=2 S=2048 D=1024 H=16 HD=64 ROT=16 FF=4096, fp32 in/out, bf16 matrix math.

typedef __attribute__((ext_vector_type(16))) __bf16 v16bf;
typedef __attribute__((ext_vector_type(8)))  float  v8f;
typedef int v4i_ __attribute__((vector_size(16)));   // builtin's b128 unit

#define B_   2
#define S_   2048
#define D_   1024
#define H_   16
#define HD_  64
#define FF_  4096
#define MROWS (B_ * S_)   // 4096 token rows

// CDNA5 async global->LDS path (guarded: falls back to sync staging if the
// toolchain doesn't declare the gfx1250 async builtins).
#if defined(__AMDGCN__) &&                                              \
    __has_builtin(__builtin_amdgcn_global_load_async_to_lds_b128) &&    \
    __has_builtin(__builtin_amdgcn_s_wait_asynccnt)
#define USE_ASYNC_LDS 1
#define ASYNC_WAIT(n) __builtin_amdgcn_s_wait_asynccnt(n)
#else
#define USE_ASYNC_LDS 0
#define ASYNC_WAIT(n)
#endif

__device__ __forceinline__ void stage_b128(const unsigned short* g,
                                           unsigned short* l) {
#if USE_ASYNC_LDS
  __builtin_amdgcn_global_load_async_to_lds_b128(
      (__attribute__((address_space(1))) v4i_*)g,
      (__attribute__((address_space(3))) v4i_*)l, 0, 0);
#else
  *(uint4*)l = *(const uint4*)g;
#endif
}

__device__ __forceinline__ unsigned short f2bf(float f) {
  unsigned u = __float_as_uint(f);
  unsigned r = u + 0x7FFFu + ((u >> 16) & 1u);   // round-to-nearest-even
  return (unsigned short)(r >> 16);
}

union Frag16 { uint4 u[2]; v16bf v; };           // 16 bf16 = 32 B

// ---------------------------------------------------------------- fp32->bf16
__global__ void f32_to_bf16_kernel(const float* __restrict__ src,
                                   unsigned short* __restrict__ dst, int n) {
  for (int i = blockIdx.x * blockDim.x + threadIdx.x; i < n;
       i += gridDim.x * blockDim.x)
    dst[i] = f2bf(src[i]);
}

// --------------------------------------------------- dual LayerNorm (1 row/WG)
__global__ __launch_bounds__(256)
void ln_dual_kernel(const float* __restrict__ x,
                    const float* __restrict__ w1, const float* __restrict__ bb1,
                    const float* __restrict__ w2, const float* __restrict__ bb2,
                    unsigned short* __restrict__ h1,
                    unsigned short* __restrict__ h2) {
  __shared__ float red[256];
  const int row = blockIdx.x;
  const int t = threadIdx.x;
  const float* xr = x + (size_t)row * D_;
  float v[4]; float s = 0.f;
  for (int i = 0; i < 4; ++i) { v[i] = xr[t + 256 * i]; s += v[i]; }
  red[t] = s; __syncthreads();
  for (int off = 128; off > 0; off >>= 1) {
    if (t < off) red[t] += red[t + off];
    __syncthreads();
  }
  const float mu = red[0] * (1.f / D_);
  __syncthreads();
  float q = 0.f;
  for (int i = 0; i < 4; ++i) { float d = v[i] - mu; q += d * d; }
  red[t] = q; __syncthreads();
  for (int off = 128; off > 0; off >>= 1) {
    if (t < off) red[t] += red[t + off];
    __syncthreads();
  }
  const float rstd = rsqrtf(red[0] * (1.f / D_) + 1e-5f);
  for (int i = 0; i < 4; ++i) {
    const int c = t + 256 * i;
    const float nz = (v[i] - mu) * rstd;
    h1[(size_t)row * D_ + c] = f2bf(nz * w1[c] + bb1[c]);
    h2[(size_t)row * D_ + c] = f2bf(nz * w2[c] + bb2[c]);
  }
}

// ------------------------------------------------------- WMMA GEMM, C = A*B^T
// A: M x K bf16 row-major. Bw: N x K bf16 row-major (torch-style W).
// Block tile 128x128 (8 waves, 32x64 per wave -> 8 WMMA / K-step).
// Double-buffered LDS, async global->LDS staging when available.
// EPI 0: fp32 out = acc + bias ; EPI 1: bf16 out = gelu(acc + bias)
template <int EPI>
__global__ __launch_bounds__(256)
void gemm_bf16_kernel(const unsigned short* __restrict__ A,
                      const unsigned short* __restrict__ Bw,
                      const float* __restrict__ bias,
                      void* __restrict__ Cout, int M, int N, int K) {
  __shared__ __align__(16) unsigned short lA[2][128 * 40];  // pitch 40 (80B)
  __shared__ __align__(16) unsigned short lB[2][128 * 40];
  const int tid = threadIdx.x;
  const int lane = tid & 31, wid = tid >> 5;
  const int wm = wid >> 1, wn = wid & 1;            // 4x2 wave grid
  const int hf = lane >> 4, l15 = lane & 15;
  const int gm0 = blockIdx.y * 128, gn0 = blockIdx.x * 128;
  v8f c[2][4] = {};
  const int row = tid >> 1, col = (tid & 1) * 16;   // 128x32 tile / 256 thr
  const unsigned short* Ar = A + (size_t)(gm0 + row) * K + col;
  const unsigned short* Br = Bw + (size_t)(gn0 + row) * K + col;
  unsigned short* la = (unsigned short*)&lA[0][row * 40 + col];
  unsigned short* lb = (unsigned short*)&lB[0][row * 40 + col];
  const int lstride = 128 * 40;                     // buffer stride (elements)

  // prologue: stage K-tile 0 into buffer 0 (4 x b128 per thread)
  stage_b128(Ar + 0, la + 0);
  stage_b128(Ar + 8, la + 8);
  stage_b128(Br + 0, lb + 0);
  stage_b128(Br + 8, lb + 8);

  for (int k0 = 0; k0 < K; k0 += 32) {
    const int cur = (k0 >> 5) & 1;
    if (k0 + 32 < K) {                        // stage next tile into buf cur^1
      const int nb = (cur ^ 1) * lstride;
      stage_b128(Ar + k0 + 32, la + nb + 0);
      stage_b128(Ar + k0 + 40, la + nb + 8);
      stage_b128(Br + k0 + 32, lb + nb + 0);
      stage_b128(Br + k0 + 40, lb + nb + 8);
      if (k0 + 64 < K) {                      // CDNA5 global_prefetch_b8
        __builtin_prefetch(Ar + k0 + 64, 0, 0);
        __builtin_prefetch(Br + k0 + 64, 0, 0);
      }
      ASYNC_WAIT(4);                          // current tile landed
    } else {
      ASYNC_WAIT(0);
    }
    __syncthreads();                          // all waves' stages visible

    // A fragments: lanes 0-15 K={0..7,16..23}; lanes 16-31 K={8..15,24..31}
    Frag16 afrag[2], bfrag[4];
    const int kb = hf ? 8 : 0;
    for (int i = 0; i < 2; ++i) {
      const int r = wm * 32 + i * 16 + l15;
      afrag[i].u[0] = *(const uint4*)(&lA[cur][r * 40 + kb]);
      afrag[i].u[1] = *(const uint4*)(&lA[cur][r * 40 + kb + 16]);
    }
    // B fragments: lanes 0-15 K={0..7,16..23}; lanes 16-31 K={8..15,24..31} (n = lane&15)
    const int kb16 = hf ? 8 : 0;
    for (int j = 0; j < 4; ++j) {
      const int n = wn * 64 + j * 16 + l15;
      bfrag[j].u[0] = *(const uint4*)(&lB[cur][n * 40 + kb16]);
      bfrag[j].u[1] = *(const uint4*)(&lB[cur][n * 40 + kb16 + 16]);
    }
    for (int i = 0; i < 2; ++i)
      for (int j = 0; j < 4; ++j)
        c[i][j] = __builtin_amdgcn_wmma_f32_16x16x32_bf16(
            false, afrag[i].v, false, bfrag[j].v, (short)0, c[i][j], false,
            false);
    __syncthreads();                          // done reading before overwrite
  }
  // epilogue: C layout -> lane holds (m = r + 8*hf, n = l15) per 16x16 tile
  for (int i = 0; i < 2; ++i)
    for (int j = 0; j < 4; ++j) {
      const int n = gn0 + wn * 64 + j * 16 + l15;
      const float bv = bias[n];
      for (int r = 0; r < 8; ++r) {
        const int m = gm0 + wm * 32 + i * 16 + r + hf * 8;
        const float val = c[i][j][r] + bv;
        if (EPI == 0) {
          ((float*)Cout)[(size_t)m * N + n] = val;
        } else {
          const float g = 0.5f * val * (1.0f + erff(val * 0.70710678118f));
          ((unsigned short*)Cout)[(size_t)m * N + n] = f2bf(g);
        }
      }
    }
}

// -------------------------------------------- RoPE + head split, fp32 -> bf16
// dst layout [B,H,S,HD]; src = qkv fp32 [B,S,3D]
__global__ void rope_split_kernel(const float* __restrict__ qkv,
                                  const float* __restrict__ fc,
                                  unsigned short* __restrict__ dst,
                                  int chunk_off, int do_rope) {
  const int idx = blockIdx.x * blockDim.x + threadIdx.x;
  if (idx >= B_ * H_ * S_ * HD_) return;
  const int d = idx & 63;
  const int s = (idx >> 6) & (S_ - 1);
  const int h = (idx >> 17) & (H_ - 1);
  const int b = idx >> 21;
  const float* src = qkv + (size_t)(b * S_ + s) * (3 * D_) + chunk_off + h * HD_;
  float val;
  if (do_rope && d < 16) {
    if (d < 8) {
      const float cv = fc[s * 16 + d * 2], sv = fc[s * 16 + d * 2 + 1];
      val = src[d] * cv - src[d + 8] * sv;
    } else {
      const int j = d - 8;
      const float cv = fc[s * 16 + j * 2], sv = fc[s * 16 + j * 2 + 1];
      val = src[j] * sv + src[d] * cv;
    }
  } else {
    val = src[d];
  }
  dst[idx] = f2bf(val);
}

// ---------------------------------------- causal flash attention, WMMA bf16
// grid (S/64, H, B); 128 threads = 4 waves, 16 q-rows per wave.
__global__ __launch_bounds__(128)
void flash_attn_kernel(const unsigned short* __restrict__ qb,
                       const unsigned short* __restrict__ kb,
                       const unsigned short* __restrict__ vb,
                       unsigned short* __restrict__ attn) {
  __shared__ __align__(16) unsigned short Kt[64 * 72];        // [key][dim]
  __shared__ __align__(16) unsigned short VtT[64 * 72];       // [dim][key]
  __shared__ __align__(16) unsigned short Pst[4 * 16 * 72];   // per-wave P
  const int qtile = blockIdx.x, h = blockIdx.y, b = blockIdx.z;
  const int tid = threadIdx.x, lane = tid & 31, wid = tid >> 5;
  const int hf = lane >> 4, l15 = lane & 15;
  const size_t headbase = (size_t)(b * H_ + h) * S_ * HD_;
  const int q0 = qtile * 64;
  const int qrw = q0 + wid * 16;

  // Q fragments (A layout), K dim 0..63 -> 2 frags
  Frag16 qf[2];
  {
    const unsigned short* qr = qb + headbase + (size_t)(qrw + l15) * HD_;
    const int kbq = hf ? 8 : 0;
    for (int f = 0; f < 2; ++f) {
      qf[f].u[0] = *(const uint4*)(qr + f * 32 + kbq);
      qf[f].u[1] = *(const uint4*)(qr + f * 32 + kbq + 16);
    }
  }
  v8f o[4] = {};
  float rowm[8], rowl[8];
  for (int r = 0; r < 8; ++r) { rowm[r] = -INFINITY; rowl[r] = 0.f; }

  const int ldrow = tid >> 1, ldc0 = (tid & 1) * 32;
  for (int kt = 0; kt <= qtile; ++kt) {
    __syncthreads();
    {   // stage K (row-major, async) and V (transposed, manual) tiles
      const unsigned short* kr =
          kb + headbase + (size_t)(kt * 64 + ldrow) * HD_ + ldc0;
      unsigned short* kl = &Kt[ldrow * 72 + ldc0];
      stage_b128(kr + 0,  kl + 0);
      stage_b128(kr + 8,  kl + 8);
      stage_b128(kr + 16, kl + 16);
      stage_b128(kr + 24, kl + 24);
      const unsigned short* vr =
          vb + headbase + (size_t)(kt * 64 + ldrow) * HD_ + ldc0;
      union { uint4 u[4]; unsigned short s[32]; } vv;
      vv.u[0] = ((const uint4*)vr)[0]; vv.u[1] = ((const uint4*)vr)[1];
      vv.u[2] = ((const uint4*)vr)[2]; vv.u[3] = ((const uint4*)vr)[3];
      for (int j = 0; j < 32; ++j) VtT[(ldc0 + j) * 72 + ldrow] = vv.s[j];
      ASYNC_WAIT(0);
    }
    __syncthreads();

    // S = Q K^T (16 q-rows x 64 keys), 8 WMMAs
    v8f sc[4] = {};
    for (int t = 0; t < 4; ++t)
      for (int ks = 0; ks < 2; ++ks) {
        Frag16 bk;
        const int n = t * 16 + l15;
        const int kb16 = (hf ? 8 : 0) + ks * 32;
        bk.u[0] = *(const uint4*)(&Kt[n * 72 + kb16]);
        bk.u[1] = *(const uint4*)(&Kt[n * 72 + kb16 + 16]);
        sc[t] = __builtin_amdgcn_wmma_f32_16x16x32_bf16(
            false, qf[ks].v, false, bk.v, (short)0, sc[t], false, false);
      }

    // online softmax (rows split across 16-lane halves, m = r + 8*hf)
    float sm[4][8];
    const bool maskT = (kt == qtile);
    for (int t = 0; t < 4; ++t) {
      const int kg = kt * 64 + t * 16 + l15;
      for (int r = 0; r < 8; ++r) {
        float s = sc[t][r] * 0.125f;   // 1/sqrt(64)
        if (maskT && kg > qrw + r + hf * 8) s = -INFINITY;
        sm[t][r] = s;
      }
    }
    float fr[8];
    for (int r = 0; r < 8; ++r) {
      float mx = sm[0][r];
      for (int t = 1; t < 4; ++t) mx = fmaxf(mx, sm[t][r]);
      for (int m = 1; m < 16; m <<= 1) mx = fmaxf(mx, __shfl_xor(mx, m, 32));
      const float mnew = fmaxf(rowm[r], mx);
      fr[r] = __expf(rowm[r] - mnew);
      rowm[r] = mnew;
      float ssum = 0.f;
      const int prow = wid * 16 * 72 + (r + hf * 8) * 72;
      for (int t = 0; t < 4; ++t) {
        const float p = __expf(sm[t][r] - mnew);
        ssum += p;
        Pst[prow + t * 16 + l15] = f2bf(p);
      }
      for (int m = 1; m < 16; m <<= 1) ssum += __shfl_xor(ssum, m, 32);
      rowl[r] = rowl[r] * fr[r] + ssum;
    }
    for (int t = 0; t < 4; ++t)
      for (int r = 0; r < 8; ++r) o[t][r] = o[t][r] * fr[r];

    // O += P V  (same-wave LDS ops are in-order; no block barrier needed)
    for (int t = 0; t < 4; ++t)
      for (int ks = 0; ks < 2; ++ks) {
        Frag16 pf, vf;
        const int kbp = (hf ? 8 : 0) + ks * 32;
        const int pbase = wid * 16 * 72 + l15 * 72;
        pf.u[0] = *(const uint4*)(&Pst[pbase + kbp]);
        pf.u[1] = *(const uint4*)(&Pst[pbase + kbp + 16]);
        const int n = t * 16 + l15;
        const int kb16 = (hf ? 8 : 0) + ks * 32;
        vf.u[0] = *(const uint4*)(&VtT[n * 72 + kb16]);
        vf.u[1] = *(const uint4*)(&VtT[n * 72 + kb16 + 16]);
        o[t] = __builtin_amdgcn_wmma_f32_16x16x32_bf16(
            false, pf.v, false, vf.v, (short)0, o[t], false, false);
      }
  }

  // write normalized output -> [B,S,H*HD] bf16 for O-projection GEMM
  for (int t = 0; t < 4; ++t)
    for (int r = 0; r < 8; ++r) {
      const int m = qrw + r + hf * 8;
      const int n = t * 16 + l15;
      attn[(size_t)(b * S_ + m) * D_ + h * HD_ + n] = f2bf(o[t][r] / rowl[r]);
    }
}

// ----------------------------------------------------------- final residual
__global__ void final_add_kernel(const float* __restrict__ x,
                                 const float* __restrict__ ff,
                                 const float* __restrict__ ao,
                                 float* __restrict__ out, int n) {
  const int i = blockIdx.x * blockDim.x + threadIdx.x;
  if (i < n) out[i] = x[i] + ff[i] + ao[i];
}

// ---------------------------------------------------------------------------
extern "C" void kernel_launch(void* const* d_in, const int* in_sizes, int n_in,
                              void* d_out, int out_size, void* d_ws,
                              size_t ws_size, hipStream_t stream) {
  const float* x     = (const float*)d_in[0];
  const float* W_qkv = (const float*)d_in[2];
  const float* b_qkv = (const float*)d_in[3];
  const float* W_o   = (const float*)d_in[4];
  const float* b_o   = (const float*)d_in[5];
  const float* ln1w  = (const float*)d_in[6];
  const float* ln1b  = (const float*)d_in[7];
  const float* ln2w  = (const float*)d_in[8];
  const float* ln2b  = (const float*)d_in[9];
  const float* W1    = (const float*)d_in[10];
  const float* b1    = (const float*)d_in[11];
  const float* W2    = (const float*)d_in[12];
  const float* b2    = (const float*)d_in[13];
  const float* fc    = (const float*)d_in[14];

  char* ws = (char*)d_ws;
  const size_t MB = 1024 * 1024;
  unsigned short* h1      = (unsigned short*)(ws + 0);        // 8 MB
  unsigned short* h2      = (unsigned short*)(ws + 8 * MB);   // 8 MB
  float*          qkv     = (float*)(ws + 16 * MB);           // 48 MB
  unsigned short* gelu    = (unsigned short*)(ws + 16 * MB);  // 32 MB (reuse)
  unsigned short* qb      = (unsigned short*)(ws + 64 * MB);  // 8 MB
  unsigned short* kbuf    = (unsigned short*)(ws + 72 * MB);  // 8 MB
  unsigned short* vbuf    = (unsigned short*)(ws + 80 * MB);  // 8 MB
  unsigned short* attn    = (unsigned short*)(ws + 88 * MB);  // 8 MB
  float*          attnout = (float*)(ws + 96 * MB);           // 16 MB
  float*          ffout   = (float*)(ws + 112 * MB);          // 16 MB
  unsigned short* wqkv_bf = (unsigned short*)(ws + 128 * MB); // 6 MB
  unsigned short* wo_bf   = (unsigned short*)(ws + 134 * MB); // 2 MB
  unsigned short* w1_bf   = (unsigned short*)(ws + 136 * MB); // 8 MB
  unsigned short* w2_bf   = (unsigned short*)(ws + 144 * MB); // 8 MB  (152 tot)

  // weights -> bf16
  f32_to_bf16_kernel<<<1024, 256, 0, stream>>>(W_qkv, wqkv_bf, 3 * D_ * D_);
  f32_to_bf16_kernel<<<1024, 256, 0, stream>>>(W_o,   wo_bf,   D_ * D_);
  f32_to_bf16_kernel<<<1024, 256, 0, stream>>>(W1,    w1_bf,   FF_ * D_);
  f32_to_bf16_kernel<<<1024, 256, 0, stream>>>(W2,    w2_bf,   D_ * FF_);

  // LN1 / LN2
  ln_dual_kernel<<<MROWS, 256, 0, stream>>>(x, ln1w, ln1b, ln2w, ln2b, h1, h2);

  // QKV projection: [4096x1024] x [1024x3072]
  gemm_bf16_kernel<0><<<dim3(3072 / 128, MROWS / 128), 256, 0, stream>>>(
      h1, wqkv_bf, b_qkv, qkv, MROWS, 3 * D_, D_);

  // RoPE + head split
  const int nqe = B_ * H_ * S_ * HD_;
  rope_split_kernel<<<nqe / 256, 256, 0, stream>>>(qkv, fc, qb,   0,       1);
  rope_split_kernel<<<nqe / 256, 256, 0, stream>>>(qkv, fc, kbuf, D_,      1);
  rope_split_kernel<<<nqe / 256, 256, 0, stream>>>(qkv, fc, vbuf, 2 * D_,  0);

  // causal flash attention
  flash_attn_kernel<<<dim3(S_ / 64, H_, B_), 128, 0, stream>>>(qb, kbuf, vbuf,
                                                               attn);
  // output projection
  gemm_bf16_kernel<0><<<dim3(D_ / 128, MROWS / 128), 256, 0, stream>>>(
      attn, wo_bf, b_o, attnout, MROWS, D_, D_);

  // FFN
  gemm_bf16_kernel<1><<<dim3(FF_ / 128, MROWS / 128), 256, 0, stream>>>(
      h2, w1_bf, b1, gelu, MROWS, FF_, D_);
  gemm_bf16_kernel<0><<<dim3(D_ / 128, MROWS / 128), 256, 0, stream>>>(
      gelu, w2_bf, b2, ffout, MROWS, D_, FF_);

  // residual sum
  const int nout = B_ * S_ * D_;
  final_add_kernel<<<(nout + 255) / 256, 256, 0, stream>>>(
      x, ffout, attnout, (float*)d_out, nout);
}
